// LocalUniFormerBlock_10239202033901
// MI455X (gfx1250) — matrix operand order repaired and checked by reading, compiled, not run on hardware
//
#include <hip/hip_runtime.h>
#include <math.h>

typedef __attribute__((ext_vector_type(16))) _Float16 v16h;
typedef __attribute__((ext_vector_type(8)))  _Float16 v8h;
typedef __attribute__((ext_vector_type(8)))  float    v8f;
typedef unsigned int u32x4 __attribute__((ext_vector_type(4)));
typedef int          i32x8 __attribute__((ext_vector_type(8)));
typedef int          i32x4 __attribute__((ext_vector_type(4)));

#define TOK_TOTAL 200704   // 1024 windows * 196 tokens
#define CCH       128
#define HIDN      512
#define THWsz     50176    // 16*56*56

// ---------------------------------------------------------------- helpers
__device__ __forceinline__ v16h cat16(v8h lo, v8h hi) {
  return __builtin_shufflevector(lo, hi, 0,1,2,3,4,5,6,7,8,9,10,11,12,13,14,15);
}
__device__ __forceinline__ v8f wmma_f16(v16h a, v16h b, v8f c) {
  return __builtin_amdgcn_wmma_f32_16x16x32_f16(false, a, false, b, (short)0, c, false, false);
}

// Tensor Data Mover: DMA a 2D tile (tile_d1 rows x tile_d0 f16 elements,
// row stride = stride_elems) from global memory into LDS at lds_off.
// D# layout per CDNA5 ISA ch.8 (group0: count/lds/global/type; group1: dims).
__device__ __forceinline__ void tdm_load_2d_f16(unsigned int lds_off, const void* gaddr,
                                                unsigned int row_len_elems,
                                                unsigned int tile_d0, unsigned int tile_d1,
                                                unsigned int stride_elems) {
  unsigned long long ga = (unsigned long long)(uintptr_t)gaddr;
  u32x4 g0;
  g0[0] = 1u;                                   // count=1 (valid user descriptor)
  g0[1] = lds_off;                              // lds_addr
  g0[2] = (unsigned int)ga;                     // global_addr[31:0]
  g0[3] = (unsigned int)(ga >> 32) | (2u << 30);// global_addr[56:32] | type=2
  i32x8 g1;
  g1[0] = (1 << 16);                            // workgroup_mask=0, data_size=1 (2B)
  g1[1] = (int)((row_len_elems & 0xFFFFu) << 16);        // tensor_dim0[15:0]
  g1[2] = (int)(row_len_elems >> 16);                    // tensor_dim0[31:16], tensor_dim1[15:0]=0
  g1[3] = (int)(16u | (tile_d0 << 16));                  // tensor_dim1=1<<20 (hi bits), tile_dim0
  g1[4] = (int)tile_d1;                                  // tile_dim1 (tile_dim2=0)
  g1[5] = (int)stride_elems;                             // tensor_dim0_stride[31:0]
  g1[6] = 0;                                             // stride[47:32]=0, dim1_stride lo=0
  g1[7] = 0;
  i32x4 g2 = {};
  i32x4 g3 = {};
#if defined(__clang_major__) && (__clang_major__ >= 23)
  i32x8 g4 = {};
  __builtin_amdgcn_tensor_load_to_lds(g0, g1, g2, g3, g4, 0);
#else
  __builtin_amdgcn_tensor_load_to_lds(g0, g1, g2, g3, 0);
#endif
}

// ---------------------------------------------------------------- fp32 -> fp16 convert
__global__ __launch_bounds__(256) void cvt_f16_kernel(const float* __restrict__ src,
                                                      _Float16* __restrict__ dst, int n) {
  int i = blockIdx.x * 256 + threadIdx.x;
  if (i < n) dst[i] = (_Float16)src[i];
}

// ---------------------------------------------------------------- BN + window partition -> f16
__global__ __launch_bounds__(256) void bn_win_kernel(
    const float* __restrict__ x, const float* __restrict__ gamma,
    const float* __restrict__ beta, const float* __restrict__ mean,
    const float* __restrict__ var, _Float16* __restrict__ win) {
  int tok = blockIdx.x * 256 + threadIdx.x;       // 0..200703 (window-order)
  int c = blockIdx.y;
  int w = tok / 196, n = tok % 196;
  int b = w >> 8, r = w & 255;
  int tI = r >> 6, hI = (r >> 3) & 7, wI = r & 7;
  int wt = n / 49, rem = n % 49, wh = rem / 7, ww = rem % 7;
  int t = tI * 4 + wt, hh = hI * 7 + wh, wwg = wI * 7 + ww;
  size_t idx = (((size_t)(b * CCH + c) * 16 + t) * 56 + hh) * 56 + wwg;
  float inv = rsqrtf(var[c] + 1e-5f);
  float xv = (x[idx] - mean[c]) * (inv * gamma[c]) + beta[c];
  win[(size_t)tok * CCH + c] = (_Float16)xv;
}

// ---------------------------------------------------------------- generic WMMA GEMM
// C[M,N] = A[M,K](f16) @ B[K,N](f16) + bias, fused epilogue.
// Block tile 128x64, 8 waves (4x2), wave tile 32x32, K-chunk 32.
// A tile staged by the Tensor Data Mover; B tile staged manually (transposed).
#define EPI_F16  0
#define EPI_GELU 1
#define EPI_PROJ 2
#define EPI_FC2  3

template <int EPI>
__global__ __launch_bounds__(256) void gemm_wmma_kernel(
    const _Float16* __restrict__ A, const _Float16* __restrict__ B,
    const float* __restrict__ bias, void* __restrict__ outp,
    const float* __restrict__ aux, int M, int N, int K) {
  __shared__ _Float16 As[128 * 32];   // [row][k]  (TDM destination)
  __shared__ _Float16 Bs[64 * 32];    // transposed: [n][k]
  const int tid = threadIdx.x, lane = tid & 31, wave = tid >> 5;
  const int mBase = blockIdx.x * 128;
  const int nBase = blockIdx.y * 64;
  const int mo = (wave & 3) * 32;
  const int no = (wave >> 2) * 32;

  v8f acc[2][2] = {};

  for (int k0 = 0; k0 < K; k0 += 32) {
    // A tile via TDM: 128 rows x 32 halves, row stride K (wave 0 issues)
    if (wave == 0) {
      tdm_load_2d_f16((unsigned int)(uintptr_t)(&As[0]),
                      A + (size_t)mBase * K + k0,
                      (unsigned int)K, 32u, 128u, (unsigned int)K);
    }
    // stage B tile transposed: 32x64 halves, one 8-half unit per thread
    {
      int oct = tid;                         // 0..255
      int kr = oct >> 3, n8 = (oct & 7) * 8;
      v8h v = *(const v8h*)(B + (size_t)(k0 + kr) * N + nBase + n8);
#pragma unroll
      for (int j = 0; j < 8; ++j) Bs[(n8 + j) * 32 + kr] = v[j];
    }
    if (wave == 0) __builtin_amdgcn_s_wait_tensorcnt(0);
    __syncthreads();

    v16h af[2], bf[2];
#pragma unroll
    for (int mi = 0; mi < 2; ++mi) {
      int r = mo + mi * 16 + (lane & 15);
      int cb = (lane < 16) ? 0 : 8;
      v8h lo = *(const v8h*)(&As[r * 32 + cb]);
      v8h hi = *(const v8h*)(&As[r * 32 + cb + 16]);
      af[mi] = cat16(lo, hi);
    }
#pragma unroll
    for (int ni = 0; ni < 2; ++ni) {
      int c = no + ni * 16 + (lane & 15);
      int kb = (lane < 16) ? 0 : 8;
      v8h lo = *(const v8h*)(&Bs[c * 32 + kb]);
      v8h hi = *(const v8h*)(&Bs[c * 32 + kb + 16]);
      bf[ni] = cat16(lo, hi);
    }
#pragma unroll
    for (int mi = 0; mi < 2; ++mi)
#pragma unroll
      for (int ni = 0; ni < 2; ++ni)
        acc[mi][ni] = wmma_f16(af[mi], bf[ni], acc[mi][ni]);
    __syncthreads();
  }

  // ---- epilogue ----
#pragma unroll
  for (int mi = 0; mi < 2; ++mi) {
#pragma unroll
    for (int ni = 0; ni < 2; ++ni) {
      int col = nBase + no + ni * 16 + (lane & 15);
      float bv = bias[col];
#pragma unroll
      for (int g = 0; g < 8; ++g) {
        int row = mBase + mo + mi * 16 + g + ((lane < 16) ? 0 : 8);
        float v = acc[mi][ni][g] + bv;
        if (EPI == EPI_F16) {
          ((_Float16*)outp)[(size_t)row * N + col] = (_Float16)v;
        } else if (EPI == EPI_GELU) {
          float gl = 0.5f * v * (1.0f + erff(v * 0.70710678118654752f));
          ((_Float16*)outp)[(size_t)row * N + col] = (_Float16)gl;
        } else if (EPI == EPI_PROJ) {
          int w = row / 196, n = row % 196;
          int b = w >> 8, r = w & 255;
          int tI = r >> 6, hI = (r >> 3) & 7, wI = r & 7;
          int wt = n / 49, rem = n % 49, wh = rem / 7, ww = rem % 7;
          int t = tI * 4 + wt, hh = hI * 7 + wh, wwg = wI * 7 + ww;
          size_t cf = (((size_t)(b * CCH + col) * 16 + t) * 56 + hh) * 56 + wwg;
          size_t cl = ((((size_t)(b * 16 + t) * 56 + hh) * 56 + wwg)) * CCH + col;
          ((float*)outp)[cl] = aux[cf] + v;
        } else {  // EPI_FC2
          int b = row / THWsz, r2 = row % THWsz;
          float o = aux[(size_t)row * CCH + col] + v;
          ((float*)outp)[((size_t)(b * CCH + col)) * THWsz + r2] = o;
        }
      }
    }
  }
}

// ---------------------------------------------------------------- windowed attention
__global__ __launch_bounds__(256) void attn_kernel(const _Float16* __restrict__ qkv,
                                                   _Float16* __restrict__ out) {
  __shared__ _Float16 kbuf[208 * 32];     // [key][d]
  __shared__ _Float16 vt[32 * 224];       // [d][key]
  __shared__ _Float16 pbuf[8][16 * 32];   // per-wave P staging
  const int tid = threadIdx.x, lane = tid & 31, wave = tid >> 5;
  const int w = blockIdx.x >> 2, h = blockIdx.x & 3;
  const size_t base = (size_t)w * 196 * 384;

  for (int oct = tid; oct < 208 * 4; oct += 256) {
    int r = oct >> 2, c8 = (oct & 3) * 8;
    v8h v = {};
    if (r < 196) v = *(const v8h*)(qkv + base + (size_t)r * 384 + 128 + h * 32 + c8);
    *(v8h*)(&kbuf[r * 32 + c8]) = v;
  }
  for (int oct = tid; oct < 224 * 4; oct += 256) {
    int r = oct >> 2, c8 = (oct & 3) * 8;
    v8h v = {};
    if (r < 196) v = *(const v8h*)(qkv + base + (size_t)r * 384 + 256 + h * 32 + c8);
#pragma unroll
    for (int j = 0; j < 8; ++j) vt[(c8 + j) * 224 + r] = v[j];
  }
  __syncthreads();

  const float scale = 0.17677669529663687f;  // 1/sqrt(32)

  for (int mt = wave; mt < 13; mt += 8) {
    v16h aq;
    {
      int qr = mt * 16 + (lane & 15);
      int cb = (lane < 16) ? 0 : 8;
      v8h lo = {}, hi = {};
      if (qr < 196) {
        lo = *(const v8h*)(qkv + base + (size_t)qr * 384 + h * 32 + cb);
        hi = *(const v8h*)(qkv + base + (size_t)qr * 384 + h * 32 + cb + 16);
      }
      aq = cat16(lo, hi);
    }
    v8f lg[13];
#pragma unroll
    for (int nt = 0; nt < 13; ++nt) {
      int key = nt * 16 + (lane & 15);
      int kb = (lane < 16) ? 0 : 8;
      v8h lo = *(const v8h*)(&kbuf[key * 32 + kb]);
      v8h hi = *(const v8h*)(&kbuf[key * 32 + kb + 16]);
      v8f z = {};
      lg[nt] = wmma_f16(aq, cat16(lo, hi), z);
    }
#pragma unroll
    for (int nt = 0; nt < 13; ++nt) lg[nt] *= scale;
    if ((lane & 15) >= 4) {
#pragma unroll
      for (int g = 0; g < 8; ++g) lg[12][g] = -1e30f;
    }
    float rinv[8];
#pragma unroll
    for (int g = 0; g < 8; ++g) {
      float m = lg[0][g];
#pragma unroll
      for (int nt = 1; nt < 13; ++nt) m = fmaxf(m, lg[nt][g]);
      m = fmaxf(m, __shfl_xor(m, 1, 32));
      m = fmaxf(m, __shfl_xor(m, 2, 32));
      m = fmaxf(m, __shfl_xor(m, 4, 32));
      m = fmaxf(m, __shfl_xor(m, 8, 32));
      float s = 0.f;
#pragma unroll
      for (int nt = 0; nt < 13; ++nt) {
        float e = __expf(lg[nt][g] - m);
        lg[nt][g] = e;
        s += e;
      }
      s += __shfl_xor(s, 1, 32);
      s += __shfl_xor(s, 2, 32);
      s += __shfl_xor(s, 4, 32);
      s += __shfl_xor(s, 8, 32);
      rinv[g] = 1.0f / s;
    }
    _Float16* pb = &pbuf[wave][0];
    v8f oac[2] = {};
    for (int kt = 0; kt < 7; ++kt) {
#pragma unroll
      for (int half = 0; half < 2; ++half) {
        int nt = kt * 2 + half;
        int pc = half * 16 + (lane & 15);
#pragma unroll
        for (int g = 0; g < 8; ++g) {
          int m = g + ((lane < 16) ? 0 : 8);
          float pv = (nt < 13) ? lg[nt][g] * rinv[g] : 0.f;
          pb[m * 32 + pc] = (_Float16)pv;
        }
      }
      int rr = lane & 15;
      int cb = (lane < 16) ? 0 : 8;
      v8h lo = *(const v8h*)(&pb[rr * 32 + cb]);
      v8h hi = *(const v8h*)(&pb[rr * 32 + cb + 16]);
      v16h ap = cat16(lo, hi);
#pragma unroll
      for (int ni = 0; ni < 2; ++ni) {
        int hd = ni * 16 + (lane & 15);
        int kb = (lane < 16) ? 0 : 8;
        v8h blo = *(const v8h*)(&vt[hd * 224 + kt * 32 + kb]);
        v8h bhi = *(const v8h*)(&vt[hd * 224 + kt * 32 + kb + 16]);
        oac[ni] = wmma_f16(ap, cat16(blo, bhi), oac[ni]);
      }
    }
#pragma unroll
    for (int ni = 0; ni < 2; ++ni) {
#pragma unroll
      for (int g = 0; g < 8; ++g) {
        int m = mt * 16 + g + ((lane < 16) ? 0 : 8);
        if (m < 196) {
          int col = h * 32 + ni * 16 + (lane & 15);
          out[((size_t)w * 196 + m) * CCH + col] = (_Float16)oac[ni][g];
        }
      }
    }
  }
}

// ---------------------------------------------------------------- LayerNorm (one wave per row)
__global__ __launch_bounds__(256) void ln_kernel(const float* __restrict__ x2,
                                                 const float* __restrict__ gamma,
                                                 const float* __restrict__ beta,
                                                 _Float16* __restrict__ out) {
  int lane = threadIdx.x & 31;
  int row = blockIdx.x * 8 + (threadIdx.x >> 5);
  const float* p = x2 + (size_t)row * CCH + lane * 4;
  float4 v = *(const float4*)p;
  float s = v.x + v.y + v.z + v.w;
  float ss = v.x * v.x + v.y * v.y + v.z * v.z + v.w * v.w;
#pragma unroll
  for (int m = 1; m <= 16; m <<= 1) {
    s += __shfl_xor(s, m, 32);
    ss += __shfl_xor(ss, m, 32);
  }
  float mean = s * (1.0f / 128.0f);
  float var = ss * (1.0f / 128.0f) - mean * mean;
  float rstd = rsqrtf(var + 1e-5f);
#pragma unroll
  for (int j = 0; j < 4; ++j) {
    int c = lane * 4 + j;
    float val = ((&v.x)[j] - mean) * rstd * gamma[c] + beta[c];
    out[(size_t)row * CCH + c] = (_Float16)val;
  }
}

// ---------------------------------------------------------------- launcher
extern "C" void kernel_launch(void* const* d_in, const int* in_sizes, int n_in,
                              void* d_out, int out_size, void* d_ws, size_t ws_size,
                              hipStream_t stream) {
  (void)in_sizes; (void)n_in; (void)out_size; (void)ws_size;
  const float* x        = (const float*)d_in[0];
  const float* bn_gamma = (const float*)d_in[1];
  const float* bn_beta  = (const float*)d_in[2];
  const float* bn_mean  = (const float*)d_in[3];
  const float* bn_var   = (const float*)d_in[4];
  const float* qkv_w    = (const float*)d_in[5];
  const float* qkv_b    = (const float*)d_in[6];
  const float* proj_w   = (const float*)d_in[7];
  const float* proj_b   = (const float*)d_in[8];
  const float* ln_gamma = (const float*)d_in[9];
  const float* ln_beta  = (const float*)d_in[10];
  const float* fc1_w    = (const float*)d_in[11];
  const float* fc1_b    = (const float*)d_in[12];
  const float* fc2_w    = (const float*)d_in[13];
  const float* fc2_b    = (const float*)d_in[14];

  char* ws = (char*)d_ws;
  _Float16* wq    = (_Float16*)(ws + 0);                        // 128*384
  _Float16* wproj = (_Float16*)(ws + 98304);                    // 128*128
  _Float16* wfc1  = (_Float16*)(ws + 131072);                   // 128*512
  _Float16* wfc2  = (_Float16*)(ws + 262144);                   // 512*128
  _Float16* winbuf  = (_Float16*)(ws + 393216);                 // 200704*128 f16 (reused: attnout)
  _Float16* qkvbuf  = (_Float16*)(ws + 393216 + 51380224ull);   // 200704*384 f16 (reused: lnbuf)
  float*    x2      = (float*)(ws + 393216 + 51380224ull + 154140672ull);          // 200704*128 f32
  _Float16* h1      = (_Float16*)(ws + 393216 + 51380224ull + 154140672ull + 102760448ull); // 200704*512 f16
  _Float16* attnout = winbuf;
  _Float16* lnbuf   = qkvbuf;

  cvt_f16_kernel<<<(128 * 384 + 255) / 256, 256, 0, stream>>>(qkv_w, wq, 128 * 384);
  cvt_f16_kernel<<<(128 * 128 + 255) / 256, 256, 0, stream>>>(proj_w, wproj, 128 * 128);
  cvt_f16_kernel<<<(128 * 512 + 255) / 256, 256, 0, stream>>>(fc1_w, wfc1, 128 * 512);
  cvt_f16_kernel<<<(512 * 128 + 255) / 256, 256, 0, stream>>>(fc2_w, wfc2, 512 * 128);

  bn_win_kernel<<<dim3(TOK_TOTAL / 256, CCH), 256, 0, stream>>>(x, bn_gamma, bn_beta,
                                                                bn_mean, bn_var, winbuf);
  gemm_wmma_kernel<EPI_F16><<<dim3(TOK_TOTAL / 128, 384 / 64), 256, 0, stream>>>(
      winbuf, wq, qkv_b, qkvbuf, nullptr, TOK_TOTAL, 384, 128);
  attn_kernel<<<1024 * 4, 256, 0, stream>>>(qkvbuf, attnout);
  gemm_wmma_kernel<EPI_PROJ><<<dim3(TOK_TOTAL / 128, 128 / 64), 256, 0, stream>>>(
      attnout, wproj, proj_b, x2, x, TOK_TOTAL, 128, 128);
  ln_kernel<<<TOK_TOTAL / 8, 256, 0, stream>>>(x2, ln_gamma, ln_beta, lnbuf);
  gemm_wmma_kernel<EPI_GELU><<<dim3(TOK_TOTAL / 128, 512 / 64), 256, 0, stream>>>(
      lnbuf, wfc1, fc1_b, h1, nullptr, TOK_TOTAL, 512, 128);
  gemm_wmma_kernel<EPI_FC2><<<dim3(TOK_TOTAL / 128, 128 / 64), 256, 0, stream>>>(
      h1, wfc2, fc2_b, d_out, x2, TOK_TOTAL, 128, 512);
}
